// SelfAttention_13881334300742
// MI455X (gfx1250) — matrix operand order repaired and checked by reading, not compiled, not run on hardware
//
#include <hip/hip_runtime.h>
#include <stdint.h>

// ---------------------------------------------------------------------------
// Types for CDNA5 WMMA
// ---------------------------------------------------------------------------
typedef __attribute__((ext_vector_type(16))) __bf16 v16bf;
typedef __attribute__((ext_vector_type(8)))  float  v8f;

union Frag16 {
    v16bf v;
    uint4 q[2];
};

// A-matrix (16x32 bf16, MxK): lanes 0-15 row M=lane hold K {0..7, 16..23};
// lanes 16-31 same rows hold K {8..15, 24..31}.
static __device__ inline v16bf load_frag_a(const char* row, int hi) {
    Frag16 f;
    f.q[0] = *(const uint4*)(row + hi * 16);
    f.q[1] = *(const uint4*)(row + hi * 16 + 32);
    return f.v;
}
// B-matrix (32x16 bf16, KxN): lane n (mod 16) holds column N=n;
// lanes 0-15 K {0..7, 16..23}, lanes 16-31 K {8..15, 24..31}.
static __device__ inline v16bf load_frag_b(const char* row, int hi) {
    Frag16 f;
    f.q[0] = *(const uint4*)(row + hi * 16);
    f.q[1] = *(const uint4*)(row + hi * 16 + 32);
    return f.v;
}

static __device__ inline v8f wmma_bf16(v16bf a, v16bf b, v8f c) {
    return __builtin_amdgcn_wmma_f32_16x16x32_bf16(
        /*neg_a=*/false, a, /*neg_b=*/false, b,
        /*c_mod=*/(short)0, c, /*reuse_a=*/false, /*reuse_b=*/false);
}

static __device__ inline unsigned short f2bf(float f) {
    unsigned u = __float_as_uint(f);
    u += 0x7FFFu + ((u >> 16) & 1u);   // round-to-nearest-even
    return (unsigned short)(u >> 16);
}

// gfx1250 async global->LDS path (guarded; fallback = load + ds_store)
#if __has_builtin(__builtin_amdgcn_global_load_async_to_lds_b128) && \
    __has_builtin(__builtin_amdgcn_s_wait_asynccnt)
#define HAVE_ASYNC_LDS 1
typedef int v4i __attribute__((__vector_size__(4 * sizeof(int))));
typedef __attribute__((address_space(3))) v4i lds_v4i;
// dst generic pointer to __shared__ -> raw 32-bit LDS offset -> AS(3) pointer
static __device__ inline void async_copy16(const void* gsrc, void* ldst) {
    __builtin_amdgcn_global_load_async_to_lds_b128(
        (v4i*)gsrc, (lds_v4i*)(unsigned int)(uintptr_t)ldst, 0, 0);
}
#else
#define HAVE_ASYNC_LDS 0
#endif

// ---------------------------------------------------------------------------
// Problem constants
// ---------------------------------------------------------------------------
#define BATCH 32
#define LSEQ  512
#define DIN   768
#define HEADS 16
#define DH    48
#define DOUT  768
#define DPAD  64          // per-head padded depth (2 x K32 WMMA chunks)
#define NEGC  1.0e12f

// ---------------------------------------------------------------------------
// Kernel 1: fp32 -> bf16 copy (activations)
// ---------------------------------------------------------------------------
__global__ void cvt_bf16_kernel(const float* __restrict__ in,
                                unsigned short* __restrict__ out, int n) {
    for (int i = blockIdx.x * blockDim.x + threadIdx.x; i < n;
         i += gridDim.x * blockDim.x)
        out[i] = f2bf(in[i]);
}

// Kernel 2: fp32 W[k][n] -> bf16 WT[n][k]
__global__ void cvt_wt_kernel(const float* __restrict__ w,
                              unsigned short* __restrict__ wt, int n) {
    for (int i = blockIdx.x * blockDim.x + threadIdx.x; i < n;
         i += gridDim.x * blockDim.x) {
        int nn = i / DIN, kk = i % DIN;
        wt[i] = f2bf(w[kk * DOUT + nn]);
    }
}

// Kernel 3: zero fill (dword granularity)
__global__ void zero_kernel(unsigned int* __restrict__ p, int nwords) {
    for (int i = blockIdx.x * blockDim.x + threadIdx.x; i < nwords;
         i += gridDim.x * blockDim.x)
        p[i] = 0u;
}

// ---------------------------------------------------------------------------
// Kernel 4: projection GEMM  C[16384 x 768] = X[16384 x 768] * W[768 x 768]
// One wave computes a 32x64 register tile (2 M x 4 N WMMA tiles): per K-chunk
// 6 fragment loads feed 8 WMMAs. K loop = 24 chunks of 32 (bf16 WMMA).
// MODE 0: write to padded [B,H,L,64]   (Q, K projections)
// MODE 1: write to transposed [B,H,48,512] (V projection)
// ---------------------------------------------------------------------------
template <int MODE>
__global__ __launch_bounds__(128) void proj_kernel(
    const unsigned short* __restrict__ X,   // [16384][768] bf16
    const unsigned short* __restrict__ WT,  // [768][768] bf16 (N-major)
    unsigned short* __restrict__ out) {
    const int lane  = threadIdx.x & 31;
    const int wave  = threadIdx.x >> 5;
    const int rh    = lane & 15;
    const int hi    = lane >> 4;

    // 512 M-groups (32 rows) x 12 N-groups (64 cols) = 6144 wave tasks
    const int task = blockIdx.x * 4 + wave;
    const int mg = task / (DOUT / 64);
    const int ng = task % (DOUT / 64);
    const int m0 = mg * 32;
    const int n0 = ng * 64;

    const char* arow0 = (const char*)(X  + (size_t)(m0 + rh) * DIN);
    const char* arow1 = (const char*)(X  + (size_t)(m0 + 16 + rh) * DIN);
    const char* brow[4];
#pragma unroll
    for (int ni = 0; ni < 4; ++ni)
        brow[ni] = (const char*)(WT + (size_t)(n0 + ni * 16 + rh) * DIN);

    v8f acc[2][4];
#pragma unroll
    for (int mi = 0; mi < 2; ++mi)
#pragma unroll
        for (int ni = 0; ni < 4; ++ni)
            acc[mi][ni] = (v8f){};

#pragma unroll 2
    for (int kc = 0; kc < DIN / 32; ++kc) {
        const int kb = kc * 64;                 // byte offset along K
        v16bf a0 = load_frag_a(arow0 + kb, hi);
        v16bf a1 = load_frag_a(arow1 + kb, hi);
#pragma unroll
        for (int ni = 0; ni < 4; ++ni) {
            v16bf b = load_frag_b(brow[ni] + kb, hi);
            acc[0][ni] = wmma_bf16(a0, b, acc[0][ni]);
            acc[1][ni] = wmma_bf16(a1, b, acc[1][ni]);
        }
    }

    // C layout: N = lane&15, M = (lane>=16 ? 8 : 0) + vgpr
#pragma unroll
    for (int ni = 0; ni < 4; ++ni) {
        const int n = n0 + ni * 16 + rh;
        const int h = n / DH;
        const int d = n % DH;
#pragma unroll
        for (int mi = 0; mi < 2; ++mi) {
#pragma unroll
            for (int r = 0; r < 8; ++r) {
                const int m  = m0 + mi * 16 + hi * 8 + r;
                const int bb = m >> 9;          // / LSEQ
                const int l  = m & (LSEQ - 1);
                size_t idx;
                if (MODE == 0)
                    idx = (((size_t)bb * HEADS + h) * LSEQ + l) * DPAD + d;
                else
                    idx = (((size_t)bb * HEADS + h) * DH + d) * LSEQ + l;
                out[idx] = f2bf(acc[mi][ni][r]);
            }
        }
    }
}

// ---------------------------------------------------------------------------
// Kernel 5: attention per (b,h). Block = 512 threads = 16 waves.
//   Stage K-head (512x64 bf16) and V^T-head (48x512 bf16) in LDS once
//   (async global->LDS when available), then loop over 16 q-tiles of 32:
//     phase1: S[k][q] = (Q_tile . K^T)*scale - masks     (64 WMMA tiles)
//     phase2: softmax over k per q column (wave32 shfl reductions)
//     phase3: O[k][d] += S_bf16 (as A, M=k,K=q) x V_tile (B, K=q,N=d)
//   final: store O with Q_len output mask.
// ---------------------------------------------------------------------------
#define S_PITCH  36          // fp32 scores row pitch (per k row, 32 q cols)
#define SB_PITCH 40          // bf16 probs row pitch

__global__ __launch_bounds__(512) void attn_kernel(
    const unsigned short* __restrict__ qp,   // [B,H,L,64] bf16
    const unsigned short* __restrict__ kp,   // [B,H,L,64] bf16
    const unsigned short* __restrict__ vt,   // [B,H,48,512] bf16
    const int* __restrict__ qlen_arr,        // [B]
    const int* __restrict__ vlen_arr,        // [B]
    float* __restrict__ out)                 // [B,L,768] fp32
{
    __shared__ float          Sld[LSEQ * S_PITCH];     // 73728 B
    __shared__ unsigned short Sb [LSEQ * SB_PITCH];    // 40960 B
    __shared__ unsigned short Kst[LSEQ * DPAD];        // 65536 B
    __shared__ unsigned short Vst[DH * LSEQ];          // 49152 B

    const int bh   = blockIdx.x;
    const int b    = bh / HEADS;
    const int h    = bh % HEADS;
    const int tid  = threadIdx.x;
    const int lane = tid & 31;
    const int wave = tid >> 5;
    const int rh   = lane & 15;
    const int hi   = lane >> 4;

    const int vlen = vlen_arr[b];
    const int qlen = qlen_arr[b];
    const float scale = 0.14433756729740643f;   // 1/sqrt(48)

    const unsigned short* qh = qp + (size_t)bh * LSEQ * DPAD;
    const unsigned short* kh = kp + (size_t)bh * LSEQ * DPAD;
    const unsigned short* vh = vt + (size_t)bh * DH * LSEQ;

    // ---- stage K and V^T heads into LDS ----
    {
        const uint4* kg = (const uint4*)kh;        // 4096 x 16B
        const uint4* vg = (const uint4*)vh;        // 3072 x 16B
        uint4* kl = (uint4*)Kst;
        uint4* vl = (uint4*)Vst;
#if HAVE_ASYNC_LDS
#pragma unroll
        for (int j = 0; j < 8; ++j)
            async_copy16(kg + tid + j * 512, kl + tid + j * 512);
#pragma unroll
        for (int j = 0; j < 6; ++j)
            async_copy16(vg + tid + j * 512, vl + tid + j * 512);
        __builtin_amdgcn_s_wait_asynccnt(0);
#else
#pragma unroll
        for (int j = 0; j < 8; ++j) kl[tid + j * 512] = kg[tid + j * 512];
#pragma unroll
        for (int j = 0; j < 6; ++j) vl[tid + j * 512] = vg[tid + j * 512];
#endif
    }
    __syncthreads();

    // wave-private O accumulators: rows [32w, 32w+32) x 48 cols
    v8f accO[2][3];
#pragma unroll
    for (int mi = 0; mi < 2; ++mi)
#pragma unroll
        for (int ni = 0; ni < 3; ++ni)
            accO[mi][ni] = (v8f){};

    for (int qt = 0; qt < LSEQ / 32; ++qt) {
        const int qbase = qt * 32;

        // ---- phase 1: scores -> Sld[k][q_local], masked + scaled ----
#pragma unroll
        for (int t = 0; t < 4; ++t) {
            const int tile = wave * 4 + t;      // 64 tiles: 2 M x 32 N
            const int mt = tile & 1;
            const int nt = tile >> 1;
            const int q0 = qbase + mt * 16;
            const int k0 = nt * 16;

            const char* arow = (const char*)(qh + (size_t)(q0 + rh) * DPAD);
            const char* brow = (const char*)&Kst[(k0 + rh) * DPAD];
            v8f acc = {};
#pragma unroll
            for (int c = 0; c < 2; ++c) {       // K = 64 = 2 x 32
                v16bf a = load_frag_a(arow + c * 64, hi);
                v16bf bb = load_frag_b(brow + c * 64, hi);
                acc = wmma_bf16(a, bb, acc);
            }
            const int kg = k0 + rh;             // this lane's key index
            const float kmask = (kg >= vlen) ? NEGC : 0.0f;
#pragma unroll
            for (int r = 0; r < 8; ++r) {
                const int qg = q0 + hi * 8 + r;
                float v = acc[r] * scale - kmask - ((kg > qg) ? NEGC : 0.0f);
                Sld[kg * S_PITCH + mt * 16 + hi * 8 + r] = v;
            }
        }
        __syncthreads();

        // ---- phase 2: softmax over k for each of the 32 q columns ----
#pragma unroll
        for (int cc = 0; cc < 2; ++cc) {
            const int qc = wave * 2 + cc;
            float m = -3.0e38f;
            for (int j = lane; j < LSEQ; j += 32)
                m = fmaxf(m, Sld[j * S_PITCH + qc]);
#pragma unroll
            for (int off = 16; off >= 1; off >>= 1)
                m = fmaxf(m, __shfl_xor(m, off, 32));
            float s = 0.0f;
            for (int j = lane; j < LSEQ; j += 32) {
                float e = __expf(Sld[j * S_PITCH + qc] - m);
                Sld[j * S_PITCH + qc] = e;
                s += e;
            }
#pragma unroll
            for (int off = 16; off >= 1; off >>= 1)
                s += __shfl_xor(s, off, 32);
            const float inv = 1.0f / fmaxf(s, 1e-30f);
            for (int j = lane; j < LSEQ; j += 32)
                Sb[j * SB_PITCH + qc] = f2bf(Sld[j * S_PITCH + qc] * inv);
        }
        __syncthreads();

        // ---- phase 3: O[k,d] += A^T V  (A = Sb, M=k, K=q_local; B = V^T) --
#pragma unroll
        for (int mi = 0; mi < 2; ++mi) {
            const int krow = wave * 32 + mi * 16 + rh;
            v16bf a = load_frag_a((const char*)&Sb[krow * SB_PITCH], hi);
#pragma unroll
            for (int ni = 0; ni < 3; ++ni) {
                const int d = ni * 16 + rh;
                const char* brow = (const char*)&Vst[d * LSEQ + qbase];
                v16bf bb = load_frag_b(brow, hi);
                accO[mi][ni] = wmma_bf16(a, bb, accO[mi][ni]);
            }
        }
        __syncthreads();   // protect Sld/Sb before next q-tile
    }

    // ---- store O with Q_len output mask ----
#pragma unroll
    for (int mi = 0; mi < 2; ++mi) {
#pragma unroll
        for (int ni = 0; ni < 3; ++ni) {
            const int dl = ni * 16 + rh;
#pragma unroll
            for (int r = 0; r < 8; ++r) {
                const int kg = wave * 32 + mi * 16 + hi * 8 + r;
                const float qm = (kg < qlen) ? 1.0f : 0.0f;
                out[((size_t)b * LSEQ + kg) * DOUT + h * DH + dl] =
                    accO[mi][ni][r] * qm;
            }
        }
    }
}

// ---------------------------------------------------------------------------
// Host launcher
// ---------------------------------------------------------------------------
extern "C" void kernel_launch(void* const* d_in, const int* in_sizes, int n_in,
                              void* d_out, int out_size, void* d_ws,
                              size_t ws_size, hipStream_t stream) {
    const float* Qs = (const float*)d_in[0];
    const float* Ks = (const float*)d_in[1];
    const float* Vs = (const float*)d_in[2];
    const float* WQ = (const float*)d_in[3];
    const float* WK = (const float*)d_in[4];
    const float* WV = (const float*)d_in[5];
    const int* Qlen = (const int*)d_in[6];
    const int* Vlen = (const int*)d_in[7];
    float* out = (float*)d_out;

    const size_t SZ_X  = (size_t)BATCH * LSEQ * DIN * 2;          // bf16
    const size_t SZ_W  = (size_t)DIN * DOUT * 2;
    const size_t SZ_QP = (size_t)BATCH * HEADS * LSEQ * DPAD * 2;

    char* w = (char*)d_ws;
    unsigned short* xq  = (unsigned short*)(w);
    unsigned short* xk  = (unsigned short*)(w + SZ_X);
    unsigned short* xv  = (unsigned short*)(w + 2 * SZ_X);
    unsigned short* wtq = (unsigned short*)(w + 3 * SZ_X);
    unsigned short* wtk = (unsigned short*)(w + 3 * SZ_X + SZ_W);
    unsigned short* wtv = (unsigned short*)(w + 3 * SZ_X + 2 * SZ_W);
    unsigned short* qp  = (unsigned short*)(w + 3 * SZ_X + 3 * SZ_W);
    unsigned short* kp  = (unsigned short*)(w + 3 * SZ_X + 3 * SZ_W + SZ_QP);
    unsigned short* vtp = (unsigned short*)(w + 3 * SZ_X + 3 * SZ_W + 2 * SZ_QP);

    const int NX = BATCH * LSEQ * DIN;      // 12,582,912
    const int NW = DIN * DOUT;              // 589,824

    cvt_bf16_kernel<<<8192, 256, 0, stream>>>(Qs, xq, NX);
    cvt_bf16_kernel<<<8192, 256, 0, stream>>>(Ks, xk, NX);
    cvt_bf16_kernel<<<8192, 256, 0, stream>>>(Vs, xv, NX);
    cvt_wt_kernel<<<2304, 256, 0, stream>>>(WQ, wtq, NW);
    cvt_wt_kernel<<<2304, 256, 0, stream>>>(WK, wtk, NW);
    cvt_wt_kernel<<<2304, 256, 0, stream>>>(WV, wtv, NW);

    // zero the padded q/k projection buffers (d = 48..63 pads must be 0)
    zero_kernel<<<8192, 256, 0, stream>>>((unsigned int*)qp, (int)(SZ_QP / 4));
    zero_kernel<<<8192, 256, 0, stream>>>((unsigned int*)kp, (int)(SZ_QP / 4));

    // projections: 512 M-groups x 12 N-groups = 6144 wave tasks / 4 per block
    proj_kernel<0><<<1536, 128, 0, stream>>>(xq, wtq, qp);
    proj_kernel<0><<<1536, 128, 0, stream>>>(xk, wtk, kp);
    proj_kernel<1><<<1536, 128, 0, stream>>>(xv, wtv, vtp);

    // attention: one block per (b, h)
    attn_kernel<<<BATCH * HEADS, 512, 0, stream>>>(qp, kp, vtp, Qlen, Vlen,
                                                   out);
}
